// GuassianContPolicyAE_35038343200963
// MI455X (gfx1250) — matrix operand order repaired and checked by reading, not compiled, not run on hardware
//
#include <hip/hip_runtime.h>
#include <hip/hip_bf16.h>
#include <math.h>

// ---------------------------------------------------------------------------
// Problem constants (match reference)
// ---------------------------------------------------------------------------
#define BSZ   4096
#define OBS   512
#define DH    1024
#define NEXP  8
#define DKK   256
#define DVV   1024
#define NTASK 10
#define OUTD  128

// ---------------------------------------------------------------------------
// Vector types (gfx1250 / CDNA5, wave32)
// ---------------------------------------------------------------------------
typedef __attribute__((ext_vector_type(16))) __bf16         v16bf;
typedef __attribute__((ext_vector_type(16))) unsigned short v16u;
typedef __attribute__((ext_vector_type(8)))  unsigned short v8u;
typedef __attribute__((ext_vector_type(4)))  unsigned short v4u;
typedef __attribute__((ext_vector_type(8)))  float          v8f;
typedef __attribute__((ext_vector_type(4)))  float          v4f;

static __device__ __forceinline__ unsigned short f2b(float f) {
    // fp32 -> bf16 bits, round-to-nearest-even
    unsigned u = __builtin_bit_cast(unsigned, f);
    unsigned r = u + 0x7FFFu + ((u >> 16) & 1u);
    return (unsigned short)(r >> 16);
}
static __device__ __forceinline__ float b2f(unsigned short u) {
    return __builtin_bit_cast(float, (unsigned)u << 16);
}

union FragU { v8u half_[2]; v16u full; };

// Async DMA: global (16B per lane) -> LDS, tracked by ASYNCcnt.
static __device__ __forceinline__ void async_load_b128(unsigned lds_byte_off,
                                                       const void* gaddr)
{
    asm volatile("global_load_async_to_lds_b128 %0, %1, off"
                 :: "v"(lds_byte_off), "v"(gaddr) : "memory");
}
static __device__ __forceinline__ void wait_async0()
{
    asm volatile("s_wait_asynccnt 0x0" ::: "memory");
}

// ---------------------------------------------------------------------------
// fp32 -> bf16 bulk converter (x, res; element count divisible by 4)
// ---------------------------------------------------------------------------
__global__ __launch_bounds__(256) void cvt_f32_bf16_kernel(
    const float* __restrict__ src, unsigned short* __restrict__ dst, long n4)
{
    long i = (long)blockIdx.x * blockDim.x + threadIdx.x;
    if (i >= n4) return;
    v4f v = ((const v4f*)src)[i];
    v4u o;
    #pragma unroll
    for (int j = 0; j < 4; ++j) o[j] = f2b(v[j]);
    ((v4u*)dst)[i] = o;
}

// ---------------------------------------------------------------------------
// Weight prep: W[K,N] fp32 -> WT[N,K] bf16, 64x64 LDS-tiled transpose.
// grid = (N/64, K/64, nmat); block = 256.
// ---------------------------------------------------------------------------
__global__ __launch_bounds__(256) void cvt_transpose_kernel(
    const float* __restrict__ src, unsigned short* __restrict__ dst,
    int K, int N)
{
    __shared__ unsigned short Ls[64][65];
    const size_t msz = (size_t)K * N;
    const float*    s = src + (size_t)blockIdx.z * msz;
    unsigned short* d = dst + (size_t)blockIdx.z * msz;
    const int kBase = blockIdx.y * 64, nBase = blockIdx.x * 64;
    const int tid = threadIdx.x;
    #pragma unroll
    for (int i = 0; i < 16; ++i) {
        int lin = tid + 256 * i;
        int r = lin >> 6, c = lin & 63;
        Ls[r][c] = f2b(s[(size_t)(kBase + r) * N + nBase + c]);
    }
    __syncthreads();
    #pragma unroll
    for (int i = 0; i < 16; ++i) {
        int lin = tid + 256 * i;
        int r = lin >> 6, c = lin & 63;
        d[(size_t)(nBase + r) * K + kBase + c] = Ls[c][r];
    }
}

// ---------------------------------------------------------------------------
// GEMM: C[M,N] = act(A[M,K](bf16) @ W + bias[N]), W given as WT[N,K] bf16.
// Block: 256 threads (8 waves). Block tile 128x128, K-step 64 (2 WMMA chunks),
// 16 v_wmma per stage. Tiles staged with global_load_async_to_lds_b128 into
// a double-buffered LDS (2 x 36 KB); ASYNCcnt + barriers overlap the next
// tile's DMA with the current tile's WMMA stream. The stage loop is unrolled
// by 2 so the buffer parity is compile-time (no v_movrels register indexing).
// Both LDS tiles are [row][k], APAD=72 row stride (144B: 16B-aligned, 36-bank
// stride -> conflict-free b128 fragment loads).
// OUT_BF16: write bf16 (chained layers) else fp32. RELU: apply max(0,.).
// Requires M%128==0, N%128==0, K%128==0 (true at all call sites).
// ---------------------------------------------------------------------------
#define BM 128
#define BN 128
#define BK 64
#define APAD 72
#define SBUF ((BM + BN) * APAD)          // elems per LDS buffer (A then B)

template <int OUT_BF16, int RELU>
__global__ __launch_bounds__(256) void gemm_kernel(
    const unsigned short* __restrict__ A, const unsigned short* __restrict__ WT,
    const float* __restrict__ bias, void* __restrict__ Cv,
    int M, int K, int N)
{
    __shared__ __align__(16) unsigned short Sh[2 * SBUF];

    const int tid  = threadIdx.x;
    const int lane = tid & 31;
    const int wid  = tid >> 5;
    const int hi   = lane >> 4;      // 0: lanes 0-15, 1: lanes 16-31
    const int l15  = lane & 15;
    const int mBase = blockIdx.y * BM;
    const int nBase = blockIdx.x * BN;
    const int wm = (wid >> 1) * 32;  // wave rows inside block tile
    const int wn = (wid & 1) * 64;   // wave cols inside block tile

    v8f acc[2][4];
    #pragma unroll
    for (int tm = 0; tm < 2; ++tm)
        #pragma unroll
        for (int tn = 0; tn < 4; ++tn) acc[tm][tn] = (v8f){};

    // per-thread staging: 4 x 16B chunks per tile side
    const int r0 = tid >> 3;                 // 0..31 (rows, step 32 per chunk)
    const int c0 = (tid & 7) * 8;            // col within 64-wide K tile
    const unsigned short* aP[4];
    const unsigned short* bP[4];
    unsigned aL[2][4], bL[2][4];             // loop-invariant LDS byte offsets
    #pragma unroll
    for (int i = 0; i < 4; ++i) {
        int r = r0 + 32 * i;
        aP[i] = A  + (size_t)(mBase + r) * K + c0;
        bP[i] = WT + (size_t)(nBase + r) * K + c0;
        #pragma unroll
        for (int d = 0; d < 2; ++d) {
            aL[d][i] = (unsigned)(unsigned long long)
                       (Sh + (size_t)d * SBUF + r * APAD + c0);
            bL[d][i] = (unsigned)(unsigned long long)
                       (Sh + (size_t)d * SBUF + BM * APAD + r * APAD + c0);
        }
    }

    // lane-constant fragment offsets (within a buffer)
    const int aoff0 = (wm + l15) * APAD + hi * 8;
    const int aoff1 = (wm + 16 + l15) * APAD + hi * 8;
    const int boff  = BM * APAD + (wn + l15) * APAD + hi * 8;

    const int steps = K / BK;                // even at all call sites

    // prologue: tile 0 -> buffer 0
    #pragma unroll
    for (int i = 0; i < 4; ++i) {
        async_load_b128(aL[0][i], aP[i]);
        async_load_b128(bL[0][i], bP[i]);
        aP[i] += BK; bP[i] += BK;
    }

    // one pipeline stage with compile-time buffer parity
#define GEMM_STAGE(CUR, DO_ISSUE)                                             \
    {                                                                         \
        wait_async0();                   /* own tile DMA complete */          \
        __syncthreads();                 /* everyone's tile is in LDS */      \
        if (DO_ISSUE) {                                                       \
            _Pragma("unroll")                                                 \
            for (int i = 0; i < 4; ++i) {                                     \
                async_load_b128(aL[(CUR) ^ 1][i], aP[i]);                     \
                async_load_b128(bL[(CUR) ^ 1][i], bP[i]);                     \
                aP[i] += BK; bP[i] += BK;                                     \
            }                                                                 \
        }                                                                     \
        const unsigned short* sbuf = Sh + (size_t)(CUR) * SBUF;               \
        _Pragma("unroll")                                                     \
        for (int kk = 0; kk < BK; kk += 32) {                                 \
            v16bf ar[2];                                                      \
            {                                                                 \
                FragU f0, f1;                                                 \
                f0.half_[0] = *(const v8u*)(sbuf + aoff0 + kk);               \
                f0.half_[1] = *(const v8u*)(sbuf + aoff0 + kk + 16);          \
                f1.half_[0] = *(const v8u*)(sbuf + aoff1 + kk);               \
                f1.half_[1] = *(const v8u*)(sbuf + aoff1 + kk + 16);          \
                ar[0] = __builtin_bit_cast(v16bf, f0.full);                   \
                ar[1] = __builtin_bit_cast(v16bf, f1.full);                   \
            }                                                                 \
            v16bf br[4];                                                      \
            _Pragma("unroll")                                                 \
            for (int tn = 0; tn < 4; ++tn) {                                  \
                FragU f;                                                      \
                f.half_[0] = *(const v8u*)(sbuf + boff + tn * 16 * APAD + kk);        \
                f.half_[1] = *(const v8u*)(sbuf + boff + tn * 16 * APAD + kk + 16);   \
                br[tn] = __builtin_bit_cast(v16bf, f.full);                   \
            }                                                                 \
            _Pragma("unroll")                                                 \
            for (int tm = 0; tm < 2; ++tm)                                    \
                _Pragma("unroll")                                             \
                for (int tn = 0; tn < 4; ++tn)                                \
                    acc[tm][tn] = __builtin_amdgcn_wmma_f32_16x16x32_bf16(    \
                        false, ar[tm], false, br[tn], (short)0, acc[tm][tn],  \
                        false, false);                                        \
        }                                                                     \
        __syncthreads();                 /* tile fully consumed */            \
    }

    for (int k = 0; k < steps; k += 2) {
        GEMM_STAGE(0, true);             // compute buf0, fill buf1
        GEMM_STAGE(1, (k + 2 < steps));  // compute buf1, fill buf0
    }
#undef GEMM_STAGE

    // ---- epilogue: C/D layout: VGPR r holds M = r + 8*hi, N = l15
    const int colBase = nBase + wn + l15;
    float bvals[4];
    #pragma unroll
    for (int tn = 0; tn < 4; ++tn) bvals[tn] = bias[colBase + tn * 16];

    #pragma unroll
    for (int tm = 0; tm < 2; ++tm) {
        #pragma unroll
        for (int r = 0; r < 8; ++r) {
            int row = mBase + wm + tm * 16 + r + 8 * hi;
            #pragma unroll
            for (int tn = 0; tn < 4; ++tn) {
                float v = acc[tm][tn][r] + bvals[tn];
                if (RELU) v = fmaxf(v, 0.0f);
                if (OUT_BF16)
                    ((unsigned short*)Cv)[(size_t)row * N + colBase + tn * 16] = f2b(v);
                else
                    ((float*)Cv)[(size_t)row * N + colBase + tn * 16] = v;
            }
        }
    }
}

// ---------------------------------------------------------------------------
// q = Wq[t][t,:] + bq[t]   (one-hot(t) @ Wq[t] selects row t)
// ---------------------------------------------------------------------------
__global__ void compute_q_kernel(const float* __restrict__ Wq,
                                 const float* __restrict__ bq,
                                 const int* __restrict__ task_id,
                                 float* __restrict__ q)
{
    int j = threadIdx.x;
    if (j < DKK) {
        int t = task_id[0];
        q[j] = Wq[((size_t)t * NTASK + t) * DKK + j] + bq[(size_t)t * DKK + j];
    }
}

// ---------------------------------------------------------------------------
// Per row b: w = dot(k[b,:], q);  res[b,:] (+)= w * v[b,:]
// v, k are bf16; q, res fp32. One wave32 per row.
// ---------------------------------------------------------------------------
template <int INIT>
__global__ __launch_bounds__(256) void attn_accum_kernel(
    const unsigned short* __restrict__ vmat, const unsigned short* __restrict__ kmat,
    const float* __restrict__ q, float* __restrict__ res)
{
    const int wid  = threadIdx.x >> 5;
    const int lane = threadIdx.x & 31;
    const int row  = blockIdx.x * 8 + wid;

    float p = 0.0f;
    #pragma unroll
    for (int j0 = 0; j0 < DKK; j0 += 32)
        p += b2f(kmat[(size_t)row * DKK + j0 + lane]) * q[j0 + lane];
    #pragma unroll
    for (int off = 16; off >= 1; off >>= 1)
        p += __shfl_xor(p, off, 32);

    #pragma unroll 4
    for (int j = lane; j < DVV; j += 32) {
        size_t idx = (size_t)row * DVV + j;
        float x = p * b2f(vmat[idx]);
        if (INIT) res[idx] = x;
        else      res[idx] += x;
    }
}

// ---------------------------------------------------------------------------
// out[:, :64] -> mean ; clip(out[:,64:]) -> log_std ; std = exp(log_std)
// d_out = [mean | std | log_std], each BSZ x 64
// ---------------------------------------------------------------------------
__global__ void finalize_kernel(const float* __restrict__ o,
                                float* __restrict__ out)
{
    const int HALF = OUTD / 2;                // 64
    const size_t TOT = (size_t)BSZ * HALF;
    size_t idx = (size_t)blockIdx.x * blockDim.x + threadIdx.x;
    if (idx >= TOT) return;
    size_t b = idx >> 6;
    int    j = (int)(idx & 63);
    float mean = o[b * OUTD + j];
    float ls   = o[b * OUTD + HALF + j];
    ls = fminf(fmaxf(ls, -20.0f), 2.0f);
    out[idx]           = mean;
    out[TOT + idx]     = expf(ls);
    out[2 * TOT + idx] = ls;
}

// ---------------------------------------------------------------------------
// Launcher
// ---------------------------------------------------------------------------
static inline void launch_cvt(const float* src, unsigned short* dst, size_t n,
                              hipStream_t stream)
{
    long n4 = (long)(n / 4);
    cvt_f32_bf16_kernel<<<(unsigned)((n4 + 255) / 256), 256, 0, stream>>>(src, dst, n4);
}
static inline void launch_cvt_tr(const float* src, unsigned short* dst,
                                 int K, int N, int nmat, hipStream_t stream)
{
    dim3 g(N / 64, K / 64, nmat);
    cvt_transpose_kernel<<<g, 256, 0, stream>>>(src, dst, K, N);
}

extern "C" void kernel_launch(void* const* d_in, const int* in_sizes, int n_in,
                              void* d_out, int out_size, void* d_ws, size_t ws_size,
                              hipStream_t stream)
{
    (void)in_sizes; (void)n_in; (void)out_size; (void)ws_size;

    const float* x    = (const float*)d_in[0];
    const int*   tid  = (const int*)  d_in[1];
    const float* Wb1  = (const float*)d_in[2];
    const float* bb1  = (const float*)d_in[3];
    const float* Wb2  = (const float*)d_in[4];
    const float* bb2  = (const float*)d_in[5];
    const float* We1  = (const float*)d_in[6];
    const float* be1  = (const float*)d_in[7];
    const float* We2  = (const float*)d_in[8];
    const float* be2  = (const float*)d_in[9];
    const float* Wv   = (const float*)d_in[10];
    const float* bv   = (const float*)d_in[11];
    const float* Wk   = (const float*)d_in[12];
    const float* bk   = (const float*)d_in[13];
    const float* Wq   = (const float*)d_in[14];
    const float* bq   = (const float*)d_in[15];
    const float* Wt1  = (const float*)d_in[16];
    const float* bt1  = (const float*)d_in[17];
    const float* Wl   = (const float*)d_in[18];
    const float* bl   = (const float*)d_in[19];

    // ---- workspace carve ----
    const size_t M4 = (size_t)BSZ * DH;        // 4M elems
    const size_t MK = (size_t)BSZ * DKK;       // 1M elems

    float* fp = (float*)d_ws;
    float* qv  = fp;                           // 256 (slot of 1024)
    float* res = fp + 1024;                    // [B,DV] fp32 attention accum
    float* ob  = res + M4;                     // [B,OUT] fp32 head output
    unsigned short* us = (unsigned short*)(ob + (size_t)BSZ * OUTD);
    unsigned short* xb   = us;                 // [B,OBS] bf16 input
    unsigned short* h1b  = xb  + (size_t)BSZ * OBS;  // [B,DH]; reused as e2
    unsigned short* hb   = h1b + M4;           // [B,DH] base features
    unsigned short* e1b  = hb  + M4;           // [B,DH]; reused as tower t
    unsigned short* vbb  = e1b + M4;           // [B,DV]
    unsigned short* kbb  = vbb + M4;           // [B,DK]
    unsigned short* resb = kbb + MK;           // [B,DV] bf16 mirror of res

    unsigned short* wt = resb + M4;            // transposed bf16 weight pool
    const size_t sWb1  = (size_t)OBS * DH;
    const size_t sDHDH = (size_t)DH * DH;
    const size_t sWe   = (size_t)NEXP * DH * DH;
    const size_t sWv   = (size_t)NEXP * DH * DVV;
    const size_t sWk   = (size_t)NEXP * DH * DKK;
    unsigned short* tWb1 = wt;
    unsigned short* tWb2 = tWb1 + sWb1;
    unsigned short* tWe1 = tWb2 + sDHDH;
    unsigned short* tWe2 = tWe1 + sWe;
    unsigned short* tWv  = tWe2 + sWe;
    unsigned short* tWk  = tWv  + sWv;
    unsigned short* tWt1 = tWk  + sWk;
    unsigned short* tWl  = tWt1 + sDHDH;

    // ---- weight prep: fp32 [K,N] -> bf16 [N,K], once per call ----
    launch_cvt_tr(Wb1, tWb1, OBS, DH,  1,    stream);
    launch_cvt_tr(Wb2, tWb2, DH,  DH,  1,    stream);
    launch_cvt_tr(We1, tWe1, DH,  DH,  NEXP, stream);
    launch_cvt_tr(We2, tWe2, DH,  DH,  NEXP, stream);
    launch_cvt_tr(Wv,  tWv,  DH,  DVV, NEXP, stream);
    launch_cvt_tr(Wk,  tWk,  DH,  DKK, NEXP, stream);
    launch_cvt_tr(Wt1, tWt1, DH,  DH,  1,    stream);
    launch_cvt_tr(Wl,  tWl,  DH,  OUTD, 1,   stream);
    launch_cvt(x, xb, (size_t)BSZ * OBS, stream);

    const dim3 blk(256);
    const dim3 gDH(DH / BN,  BSZ / BM);        // N=1024 -> 8 x 32
    const dim3 gDK(DKK / BN, BSZ / BM);        // N=256  -> 2 x 32
    const dim3 gOU(OUTD / BN, BSZ / BM);       // N=128  -> 1 x 32

    compute_q_kernel<<<1, 256, 0, stream>>>(Wq, bq, tid, qv);

    // base MLP (bf16 chained activations)
    gemm_kernel<1,1><<<gDH, blk, 0, stream>>>(xb,  tWb1, bb1, h1b, BSZ, OBS, DH);
    gemm_kernel<1,1><<<gDH, blk, 0, stream>>>(h1b, tWb2, bb2, hb,  BSZ, DH,  DH);

    // expert loop (stream-serialized; reuses e1b / h1b(e2) / vbb / kbb)
    for (int n = 0; n < NEXP; ++n) {
        const unsigned short* We1n = tWe1 + (size_t)n * DH * DH;
        const unsigned short* We2n = tWe2 + (size_t)n * DH * DH;
        const unsigned short* Wvn  = tWv  + (size_t)n * DVV * DH;  // [N,K] per expert
        const unsigned short* Wkn  = tWk  + (size_t)n * DKK * DH;
        gemm_kernel<1,1><<<gDH, blk, 0, stream>>>(hb,  We1n, be1 + (size_t)n * DH,  e1b, BSZ, DH, DH);
        gemm_kernel<1,1><<<gDH, blk, 0, stream>>>(e1b, We2n, be2 + (size_t)n * DH,  h1b, BSZ, DH, DH);
        gemm_kernel<1,0><<<gDH, blk, 0, stream>>>(h1b, Wvn,  bv  + (size_t)n * DVV, vbb, BSZ, DH, DVV);
        gemm_kernel<1,0><<<gDK, blk, 0, stream>>>(h1b, Wkn,  bk  + (size_t)n * DKK, kbb, BSZ, DH, DKK);
        if (n == 0)
            attn_accum_kernel<1><<<BSZ / 8, blk, 0, stream>>>(vbb, kbb, qv, res);
        else
            attn_accum_kernel<0><<<BSZ / 8, blk, 0, stream>>>(vbb, kbb, qv, res);
    }

    // tower + head (res must stay fp32; mirror to bf16 for the GEMM A side)
    launch_cvt(res, resb, M4, stream);
    gemm_kernel<1,1><<<gDH, blk, 0, stream>>>(resb, tWt1, bt1, e1b, BSZ, DH, DH);
    gemm_kernel<0,0><<<gOU, blk, 0, stream>>>(e1b,  tWl,  bl,  ob,  BSZ, DH, OUTD);

    // split / clip / exp -> (mean, std, log_std)
    const size_t TOT = (size_t)BSZ * (OUTD / 2);
    finalize_kernel<<<(unsigned)((TOT + 255) / 256), 256, 0, stream>>>(ob, (float*)d_out);
}
